// MultiHeadAttention_33002528703166
// MI455X (gfx1250) — matrix operand order repaired and checked by reading, not compiled, not run on hardware
//
#include <hip/hip_runtime.h>
#include <hip/hip_bf16.h>
#include <cstdint>

// ---------------------------------------------------------------------------
// MHA for MI455X (gfx1250, wave32): f32->f16 convert pass, then TDM-fed
// double-buffered WMMA GEMMs + flash attention (TDM K, async-LDS V).
// B=2, S=2048, D=1024, H=16, DK=64
// ---------------------------------------------------------------------------

#define BATCH 2
#define SEQ   2048
#define DIM   1024
#define HEADS 16
#define DK    64
#define NKS   (DIM / 32)  // 32 K-slices

typedef __attribute__((ext_vector_type(16))) _Float16 v16h;
typedef __attribute__((ext_vector_type(8)))  _Float16 v8h;
typedef __attribute__((ext_vector_type(8)))  float    v8f;
typedef __attribute__((ext_vector_type(4)))  unsigned v4u;
typedef __attribute__((ext_vector_type(8)))  int      v8i;
typedef __attribute__((ext_vector_type(4)))  int      v4i;

#if __has_builtin(__builtin_amdgcn_tensor_load_to_lds)
#define HAVE_TDM 1
#else
#define HAVE_TDM 0
#endif

// ---------------------------------------------------------------------------
// Fragment loaders (wave32 VGPR layouts per CDNA5 ISA 7.12.2).
// ---------------------------------------------------------------------------
__device__ __forceinline__ v16h frag_a16(const _Float16* __restrict__ rowp, int k0) {
  const int hi = (threadIdx.x >> 4) & 1;
  union { v16h h; unsigned u[8]; } f;
  const _Float16* p = rowp + k0 + hi * 8;
#pragma unroll
  for (int v = 0; v < 4; ++v) f.u[v]     = *(const unsigned*)(p + 2 * v);
#pragma unroll
  for (int v = 0; v < 4; ++v) f.u[4 + v] = *(const unsigned*)(p + 16 + 2 * v);
  return f.h;
}

__device__ __forceinline__ v16h frag_b16(const _Float16* __restrict__ colp, int k0) {
  const int hi = (threadIdx.x >> 4) & 1;
  union { v16h h; unsigned u[8]; } f;
  const _Float16* p = colp + k0 + hi * 8;
#pragma unroll
  for (int v = 0; v < 4; ++v) f.u[v]     = *(const unsigned*)(p + 2 * v);
#pragma unroll
  for (int v = 0; v < 4; ++v) f.u[4 + v] = *(const unsigned*)(p + 16 + 2 * v);
  return f.h;
}

__device__ __forceinline__ v8f wmma_f16(v16h a, v16h b, v8f c) {
  return __builtin_amdgcn_wmma_f32_16x16x32_f16(false, a, false, b, (short)0, c,
                                                false, false);
}

// ---------------------------------------------------------------------------
// CDNA5 data movers
// ---------------------------------------------------------------------------
__device__ __forceinline__ void async_load_lds_b128(unsigned lds_byte_addr,
                                                    unsigned glb_off,
                                                    uint64_t glb_base) {
  asm volatile("global_load_async_to_lds_b128 %0, %1, %2"
               :
               : "v"(lds_byte_addr), "v"(glb_off), "s"(glb_base)
               : "memory");
}

__device__ __forceinline__ void wait_async0() {
#if __has_builtin(__builtin_amdgcn_s_wait_asynccnt)
  __builtin_amdgcn_s_wait_asynccnt(0);
#else
  asm volatile("s_wait_asynccnt 0x0" ::: "memory");
#endif
}

__device__ __forceinline__ void wait_tensor0() {
#if __has_builtin(__builtin_amdgcn_s_wait_tensorcnt)
  __builtin_amdgcn_s_wait_tensorcnt(0);
#else
  asm volatile("s_wait_tensorcnt 0x0" ::: "memory");
#endif
}

__device__ __forceinline__ void wait_tensor2() {
#if __has_builtin(__builtin_amdgcn_s_wait_tensorcnt)
  __builtin_amdgcn_s_wait_tensorcnt(2);
#else
  asm volatile("s_wait_tensorcnt 0x2" ::: "memory");
#endif
}

#if HAVE_TDM
// 2D f16 tile DMA: rows x cols elements, global row stride = stride elements,
// LDS rows padded by pad_amount DWORDs every pad_interval DWORDs.
__device__ __forceinline__ void tdm_load_tile_f16(unsigned lds_byte_addr,
                                                  uint64_t glb_byte_addr,
                                                  int rows, int cols, int stride,
                                                  int pad_interval_code,
                                                  int pad_amount_code) {
  v4u g0 = {0x1u, lds_byte_addr, (unsigned)(glb_byte_addr & 0xffffffffull),
            (unsigned)((glb_byte_addr >> 32) & 0x1ffffffull) | 0x80000000u};
  int w0 = (1 << 16) | (1 << 20) | (pad_interval_code << 22) | (pad_amount_code << 25);
  v8i g1 = {w0,
            (cols & 0xffff) << 16,                              // tensor_dim0 lo16
            ((cols >> 16) & 0xffff) | ((rows & 0xffff) << 16),  // dim0 hi | dim1 lo
            ((rows >> 16) & 0xffff) | ((cols & 0xffff) << 16),  // dim1 hi | tile_dim0
            (rows & 0xffff),                                    // tile_dim1
            stride,                                             // dim0_stride lo32
            0, 0};
  v4i zz = {0, 0, 0, 0};
#if defined(__clang_major__) && (__clang_major__ >= 23)
  v8i zz8 = {0, 0, 0, 0, 0, 0, 0, 0};
  __builtin_amdgcn_tensor_load_to_lds(g0, g1, zz, zz, zz8, 0);
#else
  __builtin_amdgcn_tensor_load_to_lds(g0, g1, zz, zz, 0);
#endif
}
#endif

// ---------------------------------------------------------------------------
// f32 -> f16 convert (8 elements/thread)
// ---------------------------------------------------------------------------
__global__ __launch_bounds__(256) void cvt_f32_f16(const float* __restrict__ src,
                                                   _Float16* __restrict__ dst) {
  const size_t base = ((size_t)blockIdx.x * 256 + threadIdx.x) * 8;
  float4 a = *(const float4*)(src + base);
  float4 b = *(const float4*)(src + base + 4);
  v8h o;
  o[0] = (_Float16)a.x; o[1] = (_Float16)a.y; o[2] = (_Float16)a.z; o[3] = (_Float16)a.w;
  o[4] = (_Float16)b.x; o[5] = (_Float16)b.y; o[6] = (_Float16)b.z; o[7] = (_Float16)b.w;
  *(v8h*)(dst + base) = o;
}

// ---------------------------------------------------------------------------
// GEMM: Y[m,n] = sum_k X[m,k]*W[n,k] + bias[n], all-f16 operands.
// Block 256x64, 8 waves each 32(M)x64(N) -> 8 WMMAs per wave per K-slice.
// Tiles staged by TDM DMA (double-buffered, s_wait_tensorcnt pipelined).
// ---------------------------------------------------------------------------
template <bool SPLIT>
__global__ __launch_bounds__(256) void gemm_wmma_k(const _Float16* __restrict__ X,
                                                   const _Float16* __restrict__ W,
                                                   const float* __restrict__ bias,
                                                   void* __restrict__ Yv) {
  __shared__ __align__(16) _Float16 a_sm[2][256][40];  // [buf][m][k], 80B rows
  __shared__ __align__(16) _Float16 b_sm[2][64][40];   // [buf][n][k]

  const int tid  = threadIdx.x;
  const int lane = tid & 31;
  const int wid  = tid >> 5;
  const int hi   = (lane >> 4) & 1;
  const int ln   = lane & 15;
  const int m0   = blockIdx.x * 256;
  const int n0   = blockIdx.y * 64;

  v8f acc[2][4] = {};

#if HAVE_TDM
  const uint64_t xg = (uint64_t)(uintptr_t)(X + (size_t)m0 * DIM);
  const uint64_t wg = (uint64_t)(uintptr_t)(W + (size_t)n0 * DIM);
  const unsigned a_lds[2] = {(unsigned)(uintptr_t)&a_sm[0][0][0],
                             (unsigned)(uintptr_t)&a_sm[1][0][0]};
  const unsigned b_lds[2] = {(unsigned)(uintptr_t)&b_sm[0][0][0],
                             (unsigned)(uintptr_t)&b_sm[1][0][0]};
  if (wid == 0) {  // preload slice 0 (pad: +2 DW per 16 DW -> 40-half stride)
    tdm_load_tile_f16(a_lds[0], xg, 256, 32, DIM, 3, 1);
    tdm_load_tile_f16(b_lds[0], wg, 64, 32, DIM, 3, 1);
  }
#endif

  for (int kt = 0; kt < NKS; ++kt) {
    const int cur = HAVE_TDM ? (kt & 1) : 0;
#if HAVE_TDM
    __syncthreads();  // everyone done reading buf[cur^1]
    if (wid == 0) {
      if (kt + 1 < NKS) {  // DMA next slice while this one computes
        const uint64_t koff = (uint64_t)(kt + 1) * 32 * sizeof(_Float16);
        tdm_load_tile_f16(a_lds[cur ^ 1], xg + koff, 256, 32, DIM, 3, 1);
        tdm_load_tile_f16(b_lds[cur ^ 1], wg + koff, 64, 32, DIM, 3, 1);
        wait_tensor2();  // all but the 2 just-issued are complete
      } else {
        wait_tensor0();
      }
    }
    __syncthreads();  // buf[cur] published
#else
    const int k0 = kt * 32;
    __syncthreads();
#pragma unroll
    for (int i = 0; i < 16; ++i) {  // 256x32 A tile as 4096 dwords
      int idx = tid + 256 * i, row = idx >> 4, c = idx & 15;
      *(unsigned*)&a_sm[0][row][2 * c] =
          *(const unsigned*)&X[(size_t)(m0 + row) * DIM + k0 + 2 * c];
    }
#pragma unroll
    for (int i = 0; i < 4; ++i) {   // 64x32 B tile as 1024 dwords
      int idx = tid + 256 * i, row = idx >> 4, c = idx & 15;
      *(unsigned*)&b_sm[0][row][2 * c] =
          *(const unsigned*)&W[(size_t)(n0 + row) * DIM + k0 + 2 * c];
    }
    __syncthreads();
#endif

    // preload all fragments, then issue the 8-WMMA chain back-to-back
    v16h af[2], bf[4];
#pragma unroll
    for (int s = 0; s < 2; ++s) af[s] = frag_a16(&a_sm[cur][wid * 32 + s * 16 + ln][0], 0);
#pragma unroll
    for (int t = 0; t < 4; ++t) bf[t] = frag_b16(&b_sm[cur][t * 16 + ln][0], 0);
#pragma unroll
    for (int s = 0; s < 2; ++s)
#pragma unroll
      for (int t = 0; t < 4; ++t) acc[s][t] = wmma_f16(af[s], bf[t], acc[s][t]);
  }

#pragma unroll
  for (int s = 0; s < 2; ++s)
#pragma unroll
    for (int t = 0; t < 4; ++t) {
      const int n  = n0 + t * 16 + ln;
      const float bv = bias[n];
#pragma unroll
      for (int r = 0; r < 8; ++r) {
        const int m = m0 + wid * 32 + s * 16 + r + 8 * hi;
        const float val = acc[s][t][r] + bv;
        if constexpr (SPLIT) {
          const int bb = m >> 11, srow = m & (SEQ - 1);
          const int hh = n >> 6, dk = n & (DK - 1);
          ((_Float16*)Yv)[(((size_t)(bb * HEADS + hh) * SEQ) + srow) * DK + dk] =
              (_Float16)val;
        } else {
          ((float*)Yv)[(size_t)m * DIM + n] = val;
        }
      }
    }
}

// ---------------------------------------------------------------------------
// Flash attention: one block per (bh, 64-row q block); 4 waves x 16 q rows.
// K tile via TDM DMA (padded LDS rows), V tile via async global->LDS,
// prefetch of next block, batched fragment loads before each WMMA chain.
// ---------------------------------------------------------------------------
__global__ __launch_bounds__(128) void attn_kernel(const _Float16* __restrict__ Qh,
                                                   const _Float16* __restrict__ Kh,
                                                   const _Float16* __restrict__ Vh,
                                                   _Float16* __restrict__ Ah) {
  __shared__ __align__(16) _Float16 k_sm[64][72];     // [key][d], TDM-padded
  __shared__ __align__(16) _Float16 v_sm[64][64];     // [key][d], async dst
  __shared__ __align__(16) _Float16 vt_sm[64][72];    // [d][key]
  __shared__ __align__(16) _Float16 p_sm[4][16][72];  // per-wave P

  const int tid  = threadIdx.x;
  const int lane = tid & 31;
  const int wave = tid >> 5;
  const int hi   = (lane >> 4) & 1;
  const int ln   = lane & 15;
  const int qb   = blockIdx.x;
  const int bh   = blockIdx.y;
  const int b    = bh >> 4;
  const int h    = bh & (HEADS - 1);

  const _Float16* Qbase = Qh + (size_t)bh * SEQ * DK;
  const _Float16* Kbase = Kh + (size_t)bh * SEQ * DK;
  const _Float16* Vbase = Vh + (size_t)bh * SEQ * DK;

  const int qrow0 = qb * 64 + wave * 16;

  v16h qa[2];
  {
    const _Float16* qp = Qbase + (size_t)(qrow0 + ln) * DK;
    qa[0] = frag_a16(qp, 0);
    qa[1] = frag_a16(qp, 32);
  }

  const unsigned k_lds0 = (unsigned)(uintptr_t)&k_sm[0][0];
  const unsigned v_lds0 = (unsigned)(uintptr_t)&v_sm[0][0];

  v8f o[4] = {};
  float m_run[8], l_run[8];
#pragma unroll
  for (int r = 0; r < 8; ++r) { m_run[r] = -1e30f; l_run[r] = 0.0f; }

  for (int jb = 0; jb <= qb; ++jb) {
    __syncthreads();

    const _Float16* kp = Kbase + (size_t)jb * 64 * DK;
    const _Float16* vp = Vbase + (size_t)jb * 64 * DK;

    // ---- K tile: TDM DMA (pad 4 DW per 32 DW -> 72-half rows) -----------
#if HAVE_TDM
    if (wave == 0) {
      tdm_load_tile_f16(k_lds0, (uint64_t)(uintptr_t)kp, 64, 64, 64, 4, 3);
    }
#else
#pragma unroll
    for (int i = 0; i < 32; ++i) {
      int idx = tid + i * 128;
      k_sm[idx >> 6][idx & 63] = kp[idx];
    }
#endif

    // ---- V tile: async global->LDS, 512 x 16B chunks --------------------
    {
      const uint64_t vg = (uint64_t)(uintptr_t)vp;
#pragma unroll
      for (int i = 0; i < 4; ++i) {
        const unsigned off = (unsigned)((tid + i * 128) * 16);
        async_load_lds_b128(v_lds0 + off, off, vg);
      }
    }

    // prefetch next K/V block into cache while DMAs run
    if (jb + 1 <= qb) {
      if (tid < 64)       __builtin_prefetch(kp + 64 * DK + tid * DK);
      else                __builtin_prefetch(vp + 64 * DK + (tid - 64) * DK);
    }

    wait_async0();
#if HAVE_TDM
    wait_tensor0();
#endif
    __syncthreads();

    // ---- transpose V in LDS for PV B-fragments --------------------------
#pragma unroll
    for (int i = 0; i < 32; ++i) {
      int idx = tid + i * 128;
      vt_sm[idx & 63][idx >> 6] = v_sm[idx >> 6][idx & 63];
    }
    __syncthreads();

    // ---- scores: preload 8 B-frags, then 8 WMMAs back-to-back -----------
    v16h kb[8];
#pragma unroll
    for (int nt = 0; nt < 4; ++nt) {
      const _Float16* kcol = &k_sm[nt * 16 + ln][0];
      kb[2 * nt]     = frag_b16(kcol, 0);
      kb[2 * nt + 1] = frag_b16(kcol, 32);
    }
    v8f sc[4] = {};
#pragma unroll
    for (int nt = 0; nt < 4; ++nt) {
      sc[nt] = wmma_f16(qa[0], kb[2 * nt], sc[nt]);
      sc[nt] = wmma_f16(qa[1], kb[2 * nt + 1], sc[nt]);
    }

    const bool diag = (jb == qb);
#pragma unroll
    for (int nt = 0; nt < 4; ++nt) {
      const int keyg = jb * 64 + nt * 16 + ln;
#pragma unroll
      for (int r = 0; r < 8; ++r) {
        float v = sc[nt][r] * 0.125f;  // 1/sqrt(DK)
        const int qg = qrow0 + r + 8 * hi;
        if (diag && keyg > qg) v = -1e30f;  // causal mask
        sc[nt][r] = v;
      }
    }

    // ---- online softmax (row stats per 16-lane half) --------------------
#pragma unroll
    for (int r = 0; r < 8; ++r) {
      float mx = fmaxf(fmaxf(sc[0][r], sc[1][r]), fmaxf(sc[2][r], sc[3][r]));
      mx = fmaxf(mx, __shfl_xor(mx, 1, 32));
      mx = fmaxf(mx, __shfl_xor(mx, 2, 32));
      mx = fmaxf(mx, __shfl_xor(mx, 4, 32));
      mx = fmaxf(mx, __shfl_xor(mx, 8, 32));
      const float mnew = fmaxf(m_run[r], mx);
      const float corr = __expf(m_run[r] - mnew);
      m_run[r] = mnew;
      l_run[r] *= corr;
      o[0][r] *= corr; o[1][r] *= corr; o[2][r] *= corr; o[3][r] *= corr;
      float rs = 0.0f;
#pragma unroll
      for (int nt = 0; nt < 4; ++nt) {
        const float p = __expf(sc[nt][r] - mnew);
        sc[nt][r] = p;
        rs += p;
      }
      rs += __shfl_xor(rs, 1, 32);
      rs += __shfl_xor(rs, 2, 32);
      rs += __shfl_xor(rs, 4, 32);
      rs += __shfl_xor(rs, 8, 32);
      l_run[r] += rs;
    }

    // ---- stash P (C-layout -> LDS -> A-layout) --------------------------
#pragma unroll
    for (int nt = 0; nt < 4; ++nt)
#pragma unroll
      for (int r = 0; r < 8; ++r)
        p_sm[wave][r + 8 * hi][nt * 16 + ln] = (_Float16)sc[nt][r];
    __syncthreads();

    // ---- O += P * V: preload frags, then 8 WMMAs ------------------------
    v16h pa0 = frag_a16(&p_sm[wave][ln][0], 0);
    v16h pa1 = frag_a16(&p_sm[wave][ln][0], 32);
    v16h vb[8];
#pragma unroll
    for (int dt = 0; dt < 4; ++dt) {
      const _Float16* vcol = &vt_sm[dt * 16 + ln][0];
      vb[2 * dt]     = frag_b16(vcol, 0);
      vb[2 * dt + 1] = frag_b16(vcol, 32);
    }
#pragma unroll
    for (int dt = 0; dt < 4; ++dt) {
      o[dt] = wmma_f16(pa0, vb[2 * dt], o[dt]);
      o[dt] = wmma_f16(pa1, vb[2 * dt + 1], o[dt]);
    }
  }

  // ---- normalize + write merged [B, S, D] f16 ---------------------------
#pragma unroll
  for (int r = 0; r < 8; ++r) {
    const float inv = 1.0f / l_run[r];
    const int srow = qrow0 + r + 8 * hi;
    _Float16* op = Ah + ((size_t)(b * SEQ + srow)) * DIM + h * DK;
#pragma unroll
    for (int dt = 0; dt < 4; ++dt)
      op[dt * 16 + ln] = (_Float16)(o[dt][r] * inv);
  }
}

// ---------------------------------------------------------------------------
// Host launcher.  Workspace layout (f16): Qh|Kh|Vh|Ah (8MB each) + Xc (8MB,
// reused for q/k/v conversions) + Wc (2MB, reused for all 4 weights) = 42MB.
// Stream-serialized reuse keeps kernel_launch deterministic.
// ---------------------------------------------------------------------------
extern "C" void kernel_launch(void* const* d_in, const int* in_sizes, int n_in,
                              void* d_out, int out_size, void* d_ws, size_t ws_size,
                              hipStream_t stream) {
  (void)in_sizes; (void)n_in; (void)out_size; (void)ws_size;
  const float* q  = (const float*)d_in[0];
  const float* k  = (const float*)d_in[1];
  const float* v  = (const float*)d_in[2];
  const float* Wq = (const float*)d_in[4];
  const float* bq = (const float*)d_in[5];
  const float* Wk = (const float*)d_in[6];
  const float* bk = (const float*)d_in[7];
  const float* Wv = (const float*)d_in[8];
  const float* bv = (const float*)d_in[9];
  const float* Wo = (const float*)d_in[10];
  const float* bo = (const float*)d_in[11];

  const size_t BSD = (size_t)BATCH * SEQ * DIM;  // 4M elements
  const size_t WSZ = (size_t)DIM * DIM;          // 1M elements
  _Float16* Qh = (_Float16*)d_ws;
  _Float16* Kh = Qh + BSD;
  _Float16* Vh = Kh + BSD;
  _Float16* Ah = Vh + BSD;
  _Float16* Xc = Ah + BSD;   // reused activation buffer
  _Float16* Wc = Xc + BSD;   // reused weight buffer

  const dim3 cvtX((unsigned)(BSD / 8 / 256)), cvtW((unsigned)(WSZ / 8 / 256));
  const dim3 gemm_grid(BATCH * SEQ / 256, DIM / 64);
  const dim3 attn_grid(SEQ / 64, BATCH * HEADS);

  // Q projection
  hipLaunchKernelGGL(cvt_f32_f16, cvtX, dim3(256), 0, stream, q, Xc);
  hipLaunchKernelGGL(cvt_f32_f16, cvtW, dim3(256), 0, stream, Wq, Wc);
  hipLaunchKernelGGL((gemm_wmma_k<true>), gemm_grid, dim3(256), 0, stream, Xc, Wc, bq, (void*)Qh);
  // K projection
  hipLaunchKernelGGL(cvt_f32_f16, cvtX, dim3(256), 0, stream, k, Xc);
  hipLaunchKernelGGL(cvt_f32_f16, cvtW, dim3(256), 0, stream, Wk, Wc);
  hipLaunchKernelGGL((gemm_wmma_k<true>), gemm_grid, dim3(256), 0, stream, Xc, Wc, bk, (void*)Kh);
  // V projection
  hipLaunchKernelGGL(cvt_f32_f16, cvtX, dim3(256), 0, stream, v, Xc);
  hipLaunchKernelGGL(cvt_f32_f16, cvtW, dim3(256), 0, stream, Wv, Wc);
  hipLaunchKernelGGL((gemm_wmma_k<true>), gemm_grid, dim3(256), 0, stream, Xc, Wc, bv, (void*)Vh);
  // attention
  hipLaunchKernelGGL(attn_kernel, attn_grid, dim3(128), 0, stream, Qh, Kh, Vh, Ah);
  // output projection
  hipLaunchKernelGGL(cvt_f32_f16, cvtW, dim3(256), 0, stream, Wo, Wc);
  hipLaunchKernelGGL((gemm_wmma_k<false>), gemm_grid, dim3(256), 0, stream, Ah, Wc, bo, d_out);
}
